// NodeAttrAttention_63720134804088
// MI455X (gfx1250) — matrix operand order repaired and checked by reading, not compiled, not run on hardware
//
#include <hip/hip_runtime.h>

typedef __attribute__((ext_vector_type(16))) __bf16 v16bf;
typedef __attribute__((ext_vector_type(8)))  float  v8f;

#define N_NODES 20000
#define N_ATTRS 5000
#define IN_F    512
#define OUT_F   128
#define JSTRIDE 5024            // 157*32, padded attr stride for attr_hT (bf16)
#define N_TILES 157             // ceil(5000/32); tile 156 is the 8-attr tail
#define ALPHA   0.2f
#define NWAVES  4               // waves per node-block workgroup

// ---------------------------------------------------------------------------
// K0: v1 = W @ a[:128], v2 = W @ a[128:256]   (512-vectors); also zero wh2 pad
// ---------------------------------------------------------------------------
__global__ void k_wv(const float* __restrict__ W, const float* __restrict__ a,
                     float* __restrict__ v1, float* __restrict__ v2,
                     float* __restrict__ wh2) {
    int k = blockIdx.x * blockDim.x + threadIdx.x;
    if (k >= IN_F) return;
    const float* wr = W + (size_t)k * OUT_F;
    float s1 = 0.f, s2 = 0.f;
#pragma unroll 4
    for (int f = 0; f < OUT_F; ++f) {
        float w = wr[f];
        s1 += w * a[f];
        s2 += w * a[OUT_F + f];
    }
    v1[k] = s1;
    v2[k] = s2;
    if (k < JSTRIDE - N_ATTRS) wh2[N_ATTRS + k] = 0.f;   // zero pad tail
}

// ---------------------------------------------------------------------------
// K1: wh1[i] = node_emb[i] . v1   (i<20000);  wh2[j] = attr_emb[j] . v2
// One wave per row, float4 coalesced loads, xor-shuffle reduction.
// ---------------------------------------------------------------------------
__global__ void k_wh(const float* __restrict__ node, const float* __restrict__ attr,
                     const float* __restrict__ v1, const float* __restrict__ v2,
                     float* __restrict__ wh1, float* __restrict__ wh2) {
    int wave = (blockIdx.x * blockDim.x + threadIdx.x) >> 5;
    int lane = threadIdx.x & 31;
    if (wave >= N_NODES + N_ATTRS) return;
    const float* src;
    const float* v;
    if (wave < N_NODES) { src = node + (size_t)wave * IN_F;              v = v1; }
    else                { src = attr + (size_t)(wave - N_NODES) * IN_F;  v = v2; }
    float s = 0.f;
#pragma unroll
    for (int t = 0; t < IN_F / 128; ++t) {
        float4 x = *(const float4*)(src + t * 128 + lane * 4);
        float4 y = *(const float4*)(v   + t * 128 + lane * 4);
        s += x.x * y.x + x.y * y.y + x.z * y.z + x.w * y.w;
    }
#pragma unroll
    for (int off = 16; off; off >>= 1) s += __shfl_xor(s, off, 32);
    if (lane == 0) {
        if (wave < N_NODES) wh1[wave] = s;
        else                wh2[wave - N_NODES] = s;
    }
}

// ---------------------------------------------------------------------------
// K2: attr_hT[f][j] = (attr_emb @ W)[j][f] as bf16, stride JSTRIDE, pad zeroed.
// One block per attr row j: attr row broadcast, W column reads coalesced in f.
// ---------------------------------------------------------------------------
__global__ void k_attrh(const float* __restrict__ attr, const float* __restrict__ W,
                        __bf16* __restrict__ aT) {
    int j = blockIdx.x;            // 0..4999
    int f = threadIdx.x;           // 0..127
    const float* ar = attr + (size_t)j * IN_F;
    float s = 0.f;
#pragma unroll 4
    for (int k = 0; k < IN_F; ++k) s += ar[k] * W[(size_t)k * OUT_F + f];
    aT[(size_t)f * JSTRIDE + j] = (__bf16)s;
    if (j < JSTRIDE - N_ATTRS) aT[(size_t)f * JSTRIDE + N_ATTRS + j] = (__bf16)0.f;
}

// ---------------------------------------------------------------------------
// K3: one 16-node block per workgroup, 4 waves split the 157 attr tiles
// round-robin. SINGLE pass: per tile compute unnormalized p = feat?exp(e):0,
// accumulate row-sum s in f32 VALU, pack p to bf16 A operand, WMMA-accumulate
// the unnormalized numerator (bf16 exponent range makes this safe).
// Numerator + denominator reduced across waves via LDS in fixed order
// (deterministic); epilogue divides, applies ELU, stores coalesced.
// ---------------------------------------------------------------------------
__global__ __launch_bounds__(32 * NWAVES)
void k_attn(const int* __restrict__ feat, const float* __restrict__ wh1,
            const float* __restrict__ wh2, const __bf16* __restrict__ aT,
            float* __restrict__ out) {
    __shared__ float sP[NWAVES * 16];              // per-wave row sums
    __shared__ float accW[NWAVES][16 * OUT_F];     // per-wave partial numerators

    const int ib   = blockIdx.x;                   // 0..1249
    const int tid  = threadIdx.x;                  // 0..127
    const int w    = tid >> 5;                     // wave id 0..3
    const int lane = tid & 31;
    const int g    = lane >> 4;                    // lane half
    const int r    = lane & 15;                    // matrix row (A) / col (B,C)
    const int row  = ib * 16 + r;
    const int* frow = feat + (size_t)row * N_ATTRS;
    const float wh1r = wh1[row];

    v8f acc[8];
#pragma unroll
    for (int nb = 0; nb < 8; ++nb) acc[nb] = (v8f){0.f,0.f,0.f,0.f,0.f,0.f,0.f,0.f};
    float s = 0.f;

    for (int tile = w; tile < N_TILES; tile += NWAVES) {
        const int jb = tile * 32;

        // issue all 8 B-tile loads first (2x b128 each) so they overlap A-build
        const __bf16* bbase = aT + jb + g * 16;
        v16bf Bv[8];
#pragma unroll
        for (int nb = 0; nb < 8; ++nb)
            Bv[nb] = *(const v16bf*)(bbase + (size_t)(nb * 16 + r) * JSTRIDE);

        v16bf A;
        if (jb + 32 <= N_ATTRS) {
            const int j0 = jb + g * 16;
            // WGP-scope prefetch of this wave's next tile (fills all cache levels)
            __builtin_prefetch(frow + j0 + NWAVES * 32, 0, 3);
#pragma unroll
            for (int q = 0; q < 4; ++q) {
                // A-element quads: e = q*4+c ; j = j0 + q*4 + c
                const int jq = j0 + q * 4;
                int4   fq = *(const int4*)(frow + jq);
                float4 wq = *(const float4*)(wh2 + jq);
                float es, p;
                es = wh1r + wq.x; es = fmaxf(es, ALPHA * es);
                p = (fq.x > 0) ? __expf(es) : 0.f; s += p; A[q * 4 + 0] = (__bf16)p;
                es = wh1r + wq.y; es = fmaxf(es, ALPHA * es);
                p = (fq.y > 0) ? __expf(es) : 0.f; s += p; A[q * 4 + 1] = (__bf16)p;
                es = wh1r + wq.z; es = fmaxf(es, ALPHA * es);
                p = (fq.z > 0) ? __expf(es) : 0.f; s += p; A[q * 4 + 2] = (__bf16)p;
                es = wh1r + wq.w; es = fmaxf(es, ALPHA * es);
                p = (fq.w > 0) ? __expf(es) : 0.f; s += p; A[q * 4 + 3] = (__bf16)p;
            }
        } else {
            // tail tile: guarded scalar path (wave-uniform branch, EXEC stays full)
#pragma unroll
            for (int e2 = 0; e2 < 16; ++e2) {
                const int j = jb + g * 16 + e2;
                float p = 0.f;
                if (j < N_ATTRS && frow[j] > 0) {
                    float es = wh1r + wh2[j];
                    es = fmaxf(es, ALPHA * es);
                    p = __expf(es);
                }
                s += p;
                A[e2] = (__bf16)p;
            }
        }

#pragma unroll
        for (int nb = 0; nb < 8; ++nb)
            acc[nb] = __builtin_amdgcn_wmma_f32_16x16x32_bf16(
                false, A, false, Bv[nb], (short)0, acc[nb], false, false);
    }

    // ---- per-wave denominators ----
    s += __shfl_xor(s, 16, 32);                    // combine the two lane halves
    if (lane < 16) sP[w * 16 + r] = s;

    // ---- deposit partial numerators: VGPR r0 -> row r0+8g, col lane%16 ----
#pragma unroll
    for (int nb = 0; nb < 8; ++nb)
#pragma unroll
        for (int r0 = 0; r0 < 8; ++r0)
            accW[w][(r0 + 8 * g) * OUT_F + nb * 16 + r] = acc[nb][r0];

    __syncthreads();

    // ---- fixed-order cross-wave reduce, normalize, ELU, coalesced store ----
#pragma unroll
    for (int rr = 0; rr < 16; ++rr) {
        float st = sP[rr] + sP[16 + rr] + sP[32 + rr] + sP[48 + rr];
        float inv = (st > 0.f) ? 1.f / st : 0.f;
        float v = (accW[0][rr * OUT_F + tid] + accW[1][rr * OUT_F + tid]
                 + accW[2][rr * OUT_F + tid] + accW[3][rr * OUT_F + tid]) * inv;
        float o = (v > 0.f) ? v : (__expf(v) - 1.f);
        out[(size_t)(ib * 16 + rr) * OUT_F + tid] = o;
    }
}

// ---------------------------------------------------------------------------
extern "C" void kernel_launch(void* const* d_in, const int* in_sizes, int n_in,
                              void* d_out, int out_size, void* d_ws, size_t ws_size,
                              hipStream_t stream) {
    const float* node = (const float*)d_in[0];   // [20000,512]
    const float* attr = (const float*)d_in[1];   // [5000,512]
    const int*   feat = (const int*)d_in[2];     // [20000,5000]
    const float* W    = (const float*)d_in[3];   // [512,128]
    const float* a    = (const float*)d_in[4];   // [256,1]
    float*       out  = (float*)d_out;           // [20000,128]

    // workspace carve-up (hipMalloc base is 256B aligned; offsets keep 32B align)
    float*  v1  = (float*)d_ws;                  // 512
    float*  v2  = v1 + IN_F;                     // 512
    float*  wh1 = v2 + IN_F;                     // 20000
    float*  wh2 = wh1 + N_NODES;                 // JSTRIDE (5024)
    __bf16* aT  = (__bf16*)(wh2 + JSTRIDE);      // 128 * JSTRIDE bf16

    k_wv   <<<(IN_F + 255) / 256, 256, 0, stream>>>(W, a, v1, v2, wh2);
    k_wh   <<<((N_NODES + N_ATTRS) * 32 + 255) / 256, 256, 0, stream>>>(node, attr, v1, v2, wh1, wh2);
    k_attrh<<<N_ATTRS, OUT_F, 0, stream>>>(attr, W, aT);
    k_attn <<<N_NODES / 16, 32 * NWAVES, 0, stream>>>(feat, wh1, wh2, aT, out);
}
